// WeaklySelector_35150012351021
// MI455X (gfx1250) — compile-verified, matrix operand order checked by reading, not run on hardware
//
#include <hip/hip_runtime.h>
#include <hip/hip_bf16.h>
#include <math.h>

// ---------- problem constants ----------
#define B_    32
#define S_    2048
#define C_    1536
#define N_    200
#define NPAD  208       // logits stride: 13 tiles of 16
#define WROWS 256       // W padded to 16 tiles of 16 (branch-free GEMM)
#define KSEL  128
#define SCHUNK 16       // partial-sum chunks over S (2048/16 = 128 rows each)

typedef __attribute__((ext_vector_type(16))) __bf16 v16bf;
typedef __attribute__((ext_vector_type(8)))  __bf16 v8bf;
typedef __attribute__((ext_vector_type(8)))  float  v8f;

// ---------- workspace layout (bytes) ----------
#define OFF_WB     ((size_t)0)            // 256*1536*2  = 786,432
#define OFF_LOGITS ((size_t)0x100000)     // 65536*208*4 = 54,525,952
#define OFF_COL    ((size_t)0x3800000)    // 65536*4
#define OFF_MAXID  ((size_t)0x3880000)    // 32*4
#define OFF_TOPK   ((size_t)0x3890000)    // 32*128*4
#define OFF_PART   ((size_t)0x38A0000)    // 32*16*208*4 = 425,984

// load a 16-bit B fragment (two 16B halves) per wave32 WMMA layout
static __device__ __forceinline__ v16bf load_bfrag(const __bf16* p) {
    const v8bf lo = *(const v8bf*)p;
    const v8bf hi = *(const v8bf*)(p + 16);
    v16bf r;
#pragma unroll
    for (int i = 0; i < 8; ++i) { r[i] = lo[i]; r[i + 8] = hi[i]; }
    return r;
}

// =====================================================================
// K0: convert W [200,1536] f32 -> padded [256,1536] bf16 (pad rows = 0)
// =====================================================================
__global__ void k_convW(const float* __restrict__ W, __bf16* __restrict__ Wb) {
    int i = blockIdx.x * blockDim.x + threadIdx.x;       // over 256*1536
    if (i >= WROWS * C_) return;
    int row = i / C_;
    int c   = i - row * C_;
    float v = (row < N_) ? W[(size_t)row * C_ + c] : 0.0f;
    Wb[i] = (__bf16)v;
}

// =====================================================================
// K1: WMMA GEMM  logits[t, n] = x[t, :] . W[n, :] + b[n]
//     grid = 4096 M-tiles, block = 128 (4 waves).
//     Each wave: one 16-token M-tile x FOUR 16-class N-tiles, ALL valid
//     (W padded to 256 rows -> no guards, no EXEC churn around WMMA).
//     Per K-step: 4 A loads + convert once, then ALL FOUR B fragments
//     loaded up-front (one load clause) followed by 4 back-to-back WMMAs.
//     A/B 16-bit fragment layout (wave32):
//       lane%16 = row, kh = lane/16; elems0-7 = K(kk+kh*8 .. +7),
//       elems8-15 = K(kk+kh*8+16 .. +23)
// =====================================================================
__global__ void k_gemm_logits(const float* __restrict__ x,
                              const __bf16* __restrict__ Wb,
                              const float* __restrict__ bias,
                              float* __restrict__ logits) {
    const int wave   = threadIdx.x >> 5;          // 0..3
    const int lane   = threadIdx.x & 31;
    const int ntbase = wave * 4;                  // ntiles ntbase..ntbase+3
    const int mtile  = blockIdx.x;                // 16 tokens per tile
    const int row    = lane & 15;
    const int kh     = lane >> 4;

    const float* xrow = x + (size_t)(mtile * 16 + row) * C_;
    const __bf16* wrow0 = Wb + (size_t)((ntbase + 0) * 16 + row) * C_;
    const __bf16* wrow1 = Wb + (size_t)((ntbase + 1) * 16 + row) * C_;
    const __bf16* wrow2 = Wb + (size_t)((ntbase + 2) * 16 + row) * C_;
    const __bf16* wrow3 = Wb + (size_t)((ntbase + 3) * 16 + row) * C_;

    v8f acc0 = {}, acc1 = {}, acc2 = {}, acc3 = {};

    for (int kk = 0; kk < C_; kk += 32) {
        const int kb = kk + kh * 8;

        // ---- A fragment: fp32 -> bf16 in-register (shared by 4 tiles) ----
        const float4 a0 = *(const float4*)(xrow + kb);
        const float4 a1 = *(const float4*)(xrow + kb + 4);
        const float4 a2 = *(const float4*)(xrow + kb + 16);
        const float4 a3 = *(const float4*)(xrow + kb + 20);

        // ---- all 4 B fragments up-front: one load clause, then 4 WMMAs ----
        const v16bf bf0 = load_bfrag(wrow0 + kb);
        const v16bf bf1 = load_bfrag(wrow1 + kb);
        const v16bf bf2 = load_bfrag(wrow2 + kb);
        const v16bf bf3 = load_bfrag(wrow3 + kb);

        v16bf afrag;
        afrag[0]  = (__bf16)a0.x; afrag[1]  = (__bf16)a0.y;
        afrag[2]  = (__bf16)a0.z; afrag[3]  = (__bf16)a0.w;
        afrag[4]  = (__bf16)a1.x; afrag[5]  = (__bf16)a1.y;
        afrag[6]  = (__bf16)a1.z; afrag[7]  = (__bf16)a1.w;
        afrag[8]  = (__bf16)a2.x; afrag[9]  = (__bf16)a2.y;
        afrag[10] = (__bf16)a2.z; afrag[11] = (__bf16)a2.w;
        afrag[12] = (__bf16)a3.x; afrag[13] = (__bf16)a3.y;
        afrag[14] = (__bf16)a3.z; afrag[15] = (__bf16)a3.w;

        acc0 = __builtin_amdgcn_wmma_f32_16x16x32_bf16(
                false, afrag, false, bf0, (short)0, acc0, false, false);
        acc1 = __builtin_amdgcn_wmma_f32_16x16x32_bf16(
                false, afrag, false, bf1, (short)0, acc1, false, false);
        acc2 = __builtin_amdgcn_wmma_f32_16x16x32_bf16(
                false, afrag, false, bf2, (short)0, acc2, false, false);
        acc3 = __builtin_amdgcn_wmma_f32_16x16x32_bf16(
                false, afrag, false, bf3, (short)0, acc3, false, false);
    }

    // epilogue: C/D layout -> VGPR j holds row (j + 8*kh), col (lane&15)
    const v8f* accs[4] = { &acc0, &acc1, &acc2, &acc3 };
#pragma unroll
    for (int t = 0; t < 4; ++t) {
        const int n = (ntbase + t) * 16 + (lane & 15);
        if (n < N_) {
            const float bv = bias[n];
            const v8f a = *accs[t];
#pragma unroll
            for (int j = 0; j < 8; ++j) {
                const int m = j + 8 * kh;
                logits[(size_t)(mtile * 16 + m) * NPAD + n] = a[j] + bv;
            }
        }
    }
}

// =====================================================================
// K2a: partial sums of logits over S chunks.
//      grid = (32, 16); block (b, chunk) reduces 128 rows; thread t -> n=t
// =====================================================================
__global__ void k_partial_sums(const float* __restrict__ logits,
                               float* __restrict__ partials) {
    const int b     = blockIdx.x;
    const int chunk = blockIdx.y;
    const int t     = threadIdx.x;
    if (t >= N_) return;
    const int rows = S_ / SCHUNK;                 // 128
    const float* base = logits + ((size_t)b * S_ + (size_t)chunk * rows) * NPAD + t;
    float acc = 0.0f;
    for (int ss = 0; ss < rows; ++ss) acc += base[(size_t)ss * NPAD];
    partials[((size_t)b * SCHUNK + chunk) * NPAD + t] = acc;
}

// =====================================================================
// K2b: per-batch argmax of summed logits (tie -> lower index)
// =====================================================================
__global__ void k_argmax(const float* __restrict__ partials,
                         int* __restrict__ maxids) {
    const int b = blockIdx.x;
    const int t = threadIdx.x;
    float s = -INFINITY;
    if (t < N_) {
        float acc = 0.0f;
#pragma unroll
        for (int c = 0; c < SCHUNK; ++c)
            acc += partials[((size_t)b * SCHUNK + c) * NPAD + t];
        s = acc;
    }
    __shared__ float sv[256];
    __shared__ int   si[256];
    sv[t] = s; si[t] = t;
    __syncthreads();
    for (int off = 128; off > 0; off >>= 1) {
        if (t < off) {
            const bool take = (sv[t + off] > sv[t]) ||
                              (sv[t + off] == sv[t] && si[t + off] < si[t]);
            if (take) { sv[t] = sv[t + off]; si[t] = si[t + off]; }
        }
        __syncthreads();
    }
    if (t == 0) maxids[b] = si[0];
}

// =====================================================================
// K3: per-token probability of the selected class (stable softmax)
//     grid = 8192 blocks * 8 waves -> one wave per token
// =====================================================================
__global__ void k_token_col(const float* __restrict__ logits,
                            const int* __restrict__ maxids,
                            float* __restrict__ col) {
    const int wave  = threadIdx.x >> 5;
    const int lane  = threadIdx.x & 31;
    const int token = blockIdx.x * 8 + wave;
    const float* lrow = logits + (size_t)token * NPAD;

    float lv[7];
    float m = -INFINITY;
#pragma unroll
    for (int i = 0; i < 7; ++i) {
        const int n = lane + 32 * i;
        lv[i] = (n < N_) ? lrow[n] : -INFINITY;
        m = fmaxf(m, lv[i]);
    }
#pragma unroll
    for (int off = 16; off > 0; off >>= 1)
        m = fmaxf(m, __shfl_xor(m, off, 32));

    float se = 0.0f;
#pragma unroll
    for (int i = 0; i < 7; ++i) se += expf(lv[i] - m);   // exp(-inf)=0
#pragma unroll
    for (int off = 16; off > 0; off >>= 1)
        se += __shfl_xor(se, off, 32);

    if (lane == 0) {
        const int b = token >> 11;                       // token / S_
        const float lmax = lrow[maxids[b]];
        col[token] = expf(lmax - m) / se;
    }
}

// =====================================================================
// K4: per-batch full bitonic sort (2048 elems) descending, tie->lower idx;
//     emit first 128 indices. grid = 32 blocks, 1024 threads, 16 KB LDS.
// =====================================================================
__global__ void k_topk(const float* __restrict__ col, int* __restrict__ selidx) {
    __shared__ float v[S_];
    __shared__ int   id[S_];
    const int b = blockIdx.x;
    const int t = threadIdx.x;

    for (int i = t; i < S_; i += 1024) {
        v[i]  = col[(size_t)b * S_ + i];
        id[i] = i;
    }
    __syncthreads();

    for (unsigned k = 2; k <= S_; k <<= 1) {
        for (unsigned j = k >> 1; j > 0; j >>= 1) {
            for (unsigned i = t; i < S_; i += 1024) {
                const unsigned l = i ^ j;
                if (l > i) {
                    const bool before = (v[i] > v[l]) ||
                                        (v[i] == v[l] && id[i] < id[l]);
                    const bool descRegion = ((i & k) == 0);
                    if (descRegion ? !before : before) {
                        const float tv = v[i];  v[i]  = v[l];  v[l]  = tv;
                        const int   ti = id[i]; id[i] = id[l]; id[l] = ti;
                    }
                }
            }
            __syncthreads();
        }
    }
    if (t < KSEL) selidx[b * KSEL + t] = id[t];
}

// =====================================================================
// K5: gather selected feature rows: out[b, j, :] = x[b, idx[b,j], :]
//     grid = (128, 32), 256 threads, float4 coalesced
// =====================================================================
__global__ void k_gather(const float* __restrict__ x,
                         const int* __restrict__ selidx,
                         float* __restrict__ out) {
    const int j = blockIdx.x;
    const int b = blockIdx.y;
    const int src = selidx[b * KSEL + j];
    const float4* xs = (const float4*)(x   + ((size_t)b * S_ + src) * C_);
    float4*       os = (float4*)      (out + ((size_t)b * KSEL + j) * C_);
    for (int c4 = threadIdx.x; c4 < C_ / 4; c4 += blockDim.x)
        os[c4] = xs[c4];
}

// =====================================================================
extern "C" void kernel_launch(void* const* d_in, const int* in_sizes, int n_in,
                              void* d_out, int out_size, void* d_ws, size_t ws_size,
                              hipStream_t stream) {
    const float* x    = (const float*)d_in[0];   // [32, 2048, 1536]
    const float* W    = (const float*)d_in[1];   // [200, 1536]
    const float* bias = (const float*)d_in[2];   // [200]
    float*       out  = (float*)d_out;           // [32, 128, 1536]

    char* ws = (char*)d_ws;
    __bf16* Wb       = (__bf16*)(ws + OFF_WB);
    float*  logits   = (float*) (ws + OFF_LOGITS);
    float*  col      = (float*) (ws + OFF_COL);
    int*    maxids   = (int*)   (ws + OFF_MAXID);
    int*    selidx   = (int*)   (ws + OFF_TOPK);
    float*  partials = (float*) (ws + OFF_PART);

    // K0: W -> bf16 (padded to 256 rows)
    k_convW<<<(WROWS * C_ + 255) / 256, 256, 0, stream>>>(W, Wb);

    // K1: WMMA GEMM logits = x @ W^T + b  (4 N-tiles per wave, branch-free)
    k_gemm_logits<<<dim3((B_ * S_) / 16), 128, 0, stream>>>(x, Wb, bias, logits);

    // K2a/K2b: batch mean-logit argmax (two-stage, deterministic)
    k_partial_sums<<<dim3(B_, SCHUNK), 256, 0, stream>>>(logits, partials);
    k_argmax<<<B_, 256, 0, stream>>>(partials, maxids);

    // K3: per-token probability of selected class
    k_token_col<<<(B_ * S_) / 8, 256, 0, stream>>>(logits, maxids, col);

    // K4: top-128 token indices per batch (descending, tie -> lower index)
    k_topk<<<B_, 1024, 0, stream>>>(col, selidx);

    // K5: gather selected rows
    k_gather<<<dim3(KSEL, B_), 256, 0, stream>>>(x, selidx, out);
}
